// cos_layer_52398601011758
// MI455X (gfx1250) — compile-verified, matrix operand order checked by reading, not run on hardware
//
#include <hip/hip_runtime.h>

// ---------------- problem constants ----------------
#define KDIM 1024
#define KT   (KDIM / 32)        // 32 k-tiles of 32
#define NX   4096               // rows of x
#define NC   10000              // rows of centers
#define NCP  10112              // centers padded to multiple of 128
#define SCALE_F   30.0f
#define COS_EPS_F 0.86602540378443864676f   // cos(pi/6)

typedef __attribute__((ext_vector_type(16))) __bf16 v16bf;
typedef __attribute__((ext_vector_type(8)))  float  v8f;
typedef __attribute__((ext_vector_type(4)))  float  v4f;

// ---------------- async-to-LDS availability (compile-safe probes) ----------------
#if __has_builtin(__builtin_amdgcn_global_load_async_to_lds_b128)
#define HAVE_ASYNC_LDS 1
// builtin signature (from clang diagnostic): pointee is int __attribute__((vector_size(16)))
typedef int v4i_vs __attribute__((vector_size(16)));
typedef __attribute__((address_space(1))) v4i_vs global_v4i;
typedef __attribute__((address_space(3))) v4i_vs lds_v4i;
#else
#define HAVE_ASYNC_LDS 0
#endif

#if __has_builtin(__builtin_amdgcn_s_wait_asynccnt)
#define WAIT_ASYNCCNT(n) __builtin_amdgcn_s_wait_asynccnt(n)
#elif HAVE_ASYNC_LDS
#define WAIT_ASYNCCNT(n) asm volatile("s_wait_asynccnt %0" ::"n"(n) : "memory")
#else
#define WAIT_ASYNCCNT(n) ((void)0)
#endif

// ---------------- helpers ----------------
static __device__ __forceinline__ unsigned short f2bf(float f) {
  union { float f; unsigned u; } v; v.f = f;
  unsigned r = v.u + 0x7FFFu + ((v.u >> 16) & 1u);  // round-to-nearest-even
  return (unsigned short)(r >> 16);
}
static __device__ __forceinline__ float bf2f(unsigned short h) {
  union { unsigned u; float f; } v; v.u = ((unsigned)h) << 16;
  return v.f;
}

// Element index inside the packed, WMMA-register-layout array.
// 16x32 bf16 tile; ISA layout: lane L<16 holds K {0..7,16..23} of row L,
// lane L+16 holds K {8..15,24..31} of row L. 16 contiguous bf16 per lane.
static __device__ __forceinline__ size_t packed_idx(int row, int k) {
  int rt = row >> 4, m = row & 15;
  int kt = k >> 5,  kk = k & 31;
  int q = kk >> 3,  r = kk & 7;       // octet index / pos in octet
  int lane = ((q & 1) << 4) | m;
  int p    = ((q >> 1) << 3) | r;
  return ((size_t)(rt * KT + kt) * 32 + lane) * 16 + p;
}

// ---------------- kernel 1: normalize rows + split to bf16 hi/lo, packed ----------------
__global__ __launch_bounds__(256) void normalize_pack_kernel(
    const float* __restrict__ src, int nRows,
    unsigned short* __restrict__ hi, unsigned short* __restrict__ lo,
    float* __restrict__ cn_out) {
  __shared__ float red[8];
  const int row = blockIdx.x;
  const int tid = threadIdx.x;
  const int k0  = tid * 4;                 // 4 consecutive K per thread
  const size_t pidx = packed_idx(row, k0); // k0%8 in {0,4} -> 4 contiguous packed elems

  if (row >= nRows) {                      // zero padding rows (block-uniform branch)
    *(unsigned long long*)(hi + pidx) = 0ull;
    *(unsigned long long*)(lo + pidx) = 0ull;
    return;
  }

  // one-shot read of the raw input: non-temporal (don't pollute L2)
  v4f v = __builtin_nontemporal_load((const v4f*)(src + (size_t)row * KDIM + k0));
  float ss = v.x * v.x + v.y * v.y + v.z * v.z + v.w * v.w;
  #pragma unroll
  for (int off = 16; off > 0; off >>= 1) ss += __shfl_down(ss, off, 32);
  if ((tid & 31) == 0) red[tid >> 5] = ss;
  __syncthreads();
  if (tid == 0) {
    float t = 0.0f;
    #pragma unroll
    for (int i = 0; i < 8; ++i) t += red[i];
    red[0] = rsqrtf(t);
  }
  __syncthreads();
  const float rn = red[0];

  float n[4] = { v.x * rn, v.y * rn, v.z * rn, v.w * rn };
  unsigned long long ph = 0ull, pl = 0ull;
  #pragma unroll
  for (int i = 0; i < 4; ++i) {
    unsigned short h = f2bf(n[i]);
    unsigned short l = f2bf(n[i] - bf2f(h));   // residual for bf16x3 split
    ph |= ((unsigned long long)h) << (16 * i);
    pl |= ((unsigned long long)l) << (16 * i);
  }
  // packed operands are re-read many times by the GEMM: keep default (RT) stores
  *(unsigned long long*)(hi + pidx) = ph;
  *(unsigned long long*)(lo + pidx) = pl;

  if (cn_out) {  // c_n is a final output: non-temporal
    v4f o; o.x = n[0]; o.y = n[1]; o.z = n[2]; o.w = n[3];
    __builtin_nontemporal_store(o, (v4f*)(cn_out + (size_t)row * KDIM + k0));
  }
}

// ---------------- stage one 32KB k-slab (A/B hi/lo) into LDS ----------------
// stage layout (ushort units): [0,4096)=Ah  [4096,8192)=Al  [8192,12288)=Bh  [12288,16384)=Bl
static __device__ __forceinline__ void stage_tiles(
    const unsigned short* __restrict__ xh, const unsigned short* __restrict__ xl,
    const unsigned short* __restrict__ ch, const unsigned short* __restrict__ cl,
    unsigned short* stage, int tid, int mTileBase, int nTileBase, int kt) {
  #pragma unroll
  for (int i = 0; i < 2; ++i) {
    int c = tid + i * 256;            // chunk 0..511 (16B each) per region
    int t = c >> 6;                   // tile 0..7 within region
    int o = (c & 63) * 8;             // ushort offset within tile
    size_t gA = (size_t)((mTileBase + t) * KT + kt) * 512 + o;
    size_t gB = (size_t)((nTileBase + t) * KT + kt) * 512 + o;
#if HAVE_ASYNC_LDS
    __builtin_amdgcn_global_load_async_to_lds_b128(
        (global_v4i*)(xh + gA), (lds_v4i*)(stage + 0 * 4096 + c * 8), 0, 0);
    __builtin_amdgcn_global_load_async_to_lds_b128(
        (global_v4i*)(xl + gA), (lds_v4i*)(stage + 1 * 4096 + c * 8), 0, 0);
    __builtin_amdgcn_global_load_async_to_lds_b128(
        (global_v4i*)(ch + gB), (lds_v4i*)(stage + 2 * 4096 + c * 8), 0, 0);
    __builtin_amdgcn_global_load_async_to_lds_b128(
        (global_v4i*)(cl + gB), (lds_v4i*)(stage + 3 * 4096 + c * 8), 0, 0);
#else
    *(uint4*)(stage + 0 * 4096 + c * 8) = *(const uint4*)(xh + gA);
    *(uint4*)(stage + 1 * 4096 + c * 8) = *(const uint4*)(xl + gA);
    *(uint4*)(stage + 2 * 4096 + c * 8) = *(const uint4*)(ch + gB);
    *(uint4*)(stage + 3 * 4096 + c * 8) = *(const uint4*)(cl + gB);
#endif
  }
}

// ---------------- kernel 2: WMMA GEMM + epilogue ----------------
// Block tile 128(M) x 128(N), 8 waves (4 in M x 2 in N), wave tile 32x64.
// Double-buffered 2x32KB LDS; async copies tracked with ASYNCcnt.
// Per k-step: 24 v_wmma_f32_16x16x32_bf16 per wave (bf16x3 split).
__global__ __launch_bounds__(256) void cos_gemm_kernel(
    const unsigned short* __restrict__ xh, const unsigned short* __restrict__ xl,
    const unsigned short* __restrict__ ch, const unsigned short* __restrict__ cl,
    float* __restrict__ out1, float* __restrict__ out2) {
  __shared__ __align__(32) unsigned short smem[2 * 16384];  // 2 stages x 32 KB

  const int tid  = threadIdx.x;
  const int lane = tid & 31;
  const int w    = tid >> 5;   // wave 0..7
  const int wm   = w >> 1;     // 0..3 -> M quadrant (32 rows)
  const int wn   = w & 1;      // 0..1 -> N half (64 cols)
  const int mTileBase = blockIdx.x * 8;  // 16-row tiles
  const int nTileBase = blockIdx.y * 8;  // 16-col tiles

  v8f acc[2][4] = {};

  // prologue: stage 0 into buffer 0 (8 async B128 per thread)
  stage_tiles(xh, xl, ch, cl, smem, tid, mTileBase, nTileBase, 0);

  for (int kt = 0; kt < KT; ++kt) {
    unsigned short* cur = smem + ((kt & 1) ? 16384 : 0);
    unsigned short* nxt = smem + ((kt & 1) ? 0 : 16384);

    if (kt + 1 < KT) {
      stage_tiles(xh, xl, ch, cl, nxt, tid, mTileBase, nTileBase, kt + 1);
      WAIT_ASYNCCNT(8);   // next stage's 8 copies stay in flight; current stage done
    } else {
      WAIT_ASYNCCNT(0);
    }
    __syncthreads();      // current stage visible to all waves

    // ---- fragment loads (layout already WMMA-native) ----
    const unsigned short* sAh = cur;
    const unsigned short* sAl = cur + 4096;
    const unsigned short* sBh = cur + 8192;
    const unsigned short* sBl = cur + 12288;

    v16bf ah[2], al[2], bh[4], bl[4];
    #pragma unroll
    for (int mi = 0; mi < 2; ++mi) {
      int t = wm * 2 + mi;
      ah[mi] = *(const v16bf*)(sAh + t * 512 + lane * 16);
      al[mi] = *(const v16bf*)(sAl + t * 512 + lane * 16);
    }
    #pragma unroll
    for (int ni = 0; ni < 4; ++ni) {
      int t = wn * 4 + ni;
      bh[ni] = *(const v16bf*)(sBh + t * 512 + lane * 16);
      bl[ni] = *(const v16bf*)(sBl + t * 512 + lane * 16);
    }

    // ---- bf16x3: hi*hi + hi*lo + lo*hi, f32 accumulate ----
    #pragma unroll
    for (int mi = 0; mi < 2; ++mi) {
      #pragma unroll
      for (int ni = 0; ni < 4; ++ni) {
        acc[mi][ni] = __builtin_amdgcn_wmma_f32_16x16x32_bf16(
            false, ah[mi], false, bh[ni], (short)0, acc[mi][ni], false, false);
        acc[mi][ni] = __builtin_amdgcn_wmma_f32_16x16x32_bf16(
            false, ah[mi], false, bl[ni], (short)0, acc[mi][ni], false, false);
        acc[mi][ni] = __builtin_amdgcn_wmma_f32_16x16x32_bf16(
            false, al[mi], false, bh[ni], (short)0, acc[mi][ni], false, false);
      }
    }
    __syncthreads();      // all waves done reading cur before it is re-staged
  }

  // ---- epilogue: C/D layout: VGPR j -> M = j + 8*(lane>=16), N = lane&15 ----
  // outputs are write-once: non-temporal stores keep packed operands resident in L2
  const int colInLane = lane & 15;
  const int rowHalf   = (lane >> 4) * 8;
  #pragma unroll
  for (int mi = 0; mi < 2; ++mi) {
    #pragma unroll
    for (int ni = 0; ni < 4; ++ni) {
      int col  = (nTileBase + wn * 4 + ni) * 16 + colInLane;
      int row0 = (mTileBase + wm * 2 + mi) * 16 + rowHalf;
      if (col < NC) {
        #pragma unroll
        for (int j = 0; j < 8; ++j) {
          float cv = acc[mi][ni][j];
          size_t idx = (size_t)(row0 + j) * NC + col;
          __builtin_nontemporal_store(SCALE_F * (cv - 1.0f),     out1 + idx);
          __builtin_nontemporal_store(SCALE_F * (cv - COS_EPS_F), out2 + idx);
        }
      }
    }
  }
}

// ---------------- host launcher ----------------
extern "C" void kernel_launch(void* const* d_in, const int* in_sizes, int n_in,
                              void* d_out, int out_size, void* d_ws, size_t ws_size,
                              hipStream_t stream) {
  (void)in_sizes; (void)n_in; (void)out_size; (void)ws_size;
  const float* x       = (const float*)d_in[0];
  const float* centers = (const float*)d_in[1];
  float* out = (float*)d_out;

  // workspace layout (bf16 elements): xh | xl | ch | cl  (~58.2 MB total)
  unsigned short* ws = (unsigned short*)d_ws;
  unsigned short* xh = ws;
  unsigned short* xl = xh + (size_t)NX  * KDIM;
  unsigned short* ch = xl + (size_t)NX  * KDIM;
  unsigned short* cl = ch + (size_t)NCP * KDIM;

  float* out1 = out;
  float* out2 = out + (size_t)NX * NC;
  float* cn   = out + (size_t)2 * NX * NC;

  normalize_pack_kernel<<<dim3(NX),  dim3(256), 0, stream>>>(x,       NX, xh, xl, (float*)nullptr);
  normalize_pack_kernel<<<dim3(NCP), dim3(256), 0, stream>>>(centers, NC, ch, cl, cn);
  cos_gemm_kernel<<<dim3(NX / 128, NCP / 128), dim3(256), 0, stream>>>(xh, xl, ch, cl, out1, out2);
}
